// ComplexGCN_40802189312752
// MI455X (gfx1250) — matrix operand order repaired and checked by reading, not compiled, not run on hardware
//
#include <hip/hip_runtime.h>
#include <hip/hip_bf16.h>

// ---------------------------------------------------------------------------
// Types
// ---------------------------------------------------------------------------
typedef __attribute__((ext_vector_type(16))) __bf16 bfx16;
typedef __attribute__((ext_vector_type(8)))  __bf16 bfx8;
typedef __attribute__((ext_vector_type(8)))  float  fx8;

#define MODE_F32  0
#define MODE_SIG  1
#define MODE_BF16 2

static inline int cdiv_i(long long a, long long b) { return (int)((a + b - 1) / b); }

// ---------------------------------------------------------------------------
// Generic WMMA GEMM:  C[M,Nc] = A[M,K] @ Bm[Nc,K]^T   (bf16 in, f32 acc)
// Block = 256 threads (8 waves). Tile = 128(M) x 128(N), K staged in 32-chunks.
// Waves arranged 4(M) x 2(N); each wave owns a 32x64 strip -> 8 accumulators,
// so each B fragment is reused across 2 A fragments (8 WMMA per 12 ds_b128).
// Tiles are staged global->LDS with CDNA5 async copies (ASYNCcnt).
// ---------------------------------------------------------------------------
__global__ __launch_bounds__(256) void wmma_gemm_nt(
    const __bf16* __restrict__ A,    // [M,K] row-major, K % 32 == 0
    const __bf16* __restrict__ Bm,   // [Nc,K] row-major
    float* __restrict__ Cf, __bf16* __restrict__ Cb,
    int M, int Nc, int K, int ldc, int mode)
{
    __shared__ __bf16 As[128 * 32];
    __shared__ __bf16 Bs[128 * 32];

    const int tid  = threadIdx.x;
    const int lane = tid & 31;
    const int wid  = tid >> 5;
    const int wm   = wid >> 1;     // 0..3 : 32-row strip
    const int wn   = wid & 1;      // 0..1 : 64-col strip
    const int half = lane >> 4;    // 0: lanes 0-15, 1: lanes 16-31
    const int lr   = lane & 15;
    const int mBase = blockIdx.y * 128;
    const int nBase = blockIdx.x * 128;

    fx8 acc[2][4] = {};

    const int nk = K >> 5;
    for (int kt = 0; kt < nk; ++kt) {
        const int k0 = kt << 5;

        // ---- stage A & B tiles (128 rows x 32 k each) via async global->LDS ----
        #pragma unroll
        for (int i = 0; i < 2; ++i) {
            const int l  = tid + (i << 8);     // 0..511
            const int r  = l >> 2;             // 0..127
            const int kc = (l & 3) << 3;       // 0,8,16,24
            {   // A tile
                __bf16* dst = As + r * 32 + kc;
                const unsigned ldsOff = (unsigned)(unsigned long long)dst;
                if (mBase + r < M) {
                    const unsigned long long ga =
                        (unsigned long long)(A + (long long)(mBase + r) * K + k0 + kc);
                    asm volatile("global_load_async_to_lds_b128 %0, %1, off"
                                 :: "v"(ldsOff), "v"(ga) : "memory");
                } else {
                    bfx8 z = {};
                    *(bfx8*)dst = z;
                }
            }
            {   // B tile
                __bf16* dst = Bs + r * 32 + kc;
                const unsigned ldsOff = (unsigned)(unsigned long long)dst;
                if (nBase + r < Nc) {
                    const unsigned long long ga =
                        (unsigned long long)(Bm + (long long)(nBase + r) * K + k0 + kc);
                    asm volatile("global_load_async_to_lds_b128 %0, %1, off"
                                 :: "v"(ldsOff), "v"(ga) : "memory");
                } else {
                    bfx8 z = {};
                    *(bfx8*)dst = z;
                }
            }
        }
        // prefetch next K-chunk toward L2 (global_prefetch_b8)
        if (kt + 1 < nk) {
            const int r  = tid >> 2;
            const int kc = (tid & 3) << 3;
            if (mBase + r < M)  __builtin_prefetch(A  + (long long)(mBase + r) * K + k0 + 32 + kc, 0, 1);
            if (nBase + r < Nc) __builtin_prefetch(Bm + (long long)(nBase + r) * K + k0 + 32 + kc, 0, 1);
        }
        asm volatile("s_wait_asynccnt 0" ::: "memory");
        __syncthreads();

        // ---- A fragments (16x32 bf16 layout):
        //   lane<16 : row=lr, K in {0..7, 16..23}; lane>=16: row=lr, K in {8..15, 24..31}
        union { bfx16 v; bfx8 h[2]; } af[2];
        #pragma unroll
        for (int mi = 0; mi < 2; ++mi) {
            const __bf16* p = As + (wm * 32 + mi * 16 + lr) * 32;
            af[mi].h[0] = *(const bfx8*)(p + half * 8);
            af[mi].h[1] = *(const bfx8*)(p + 16 + half * 8);
        }

        // ---- B fragments (32x16): lane holds column n=lr, K in {half*8..half*8+7, 16+half*8..16+half*8+7}
        #pragma unroll
        for (int j = 0; j < 4; ++j) {
            union { bfx16 v; bfx8 h[2]; } bf;
            const __bf16* p = Bs + (wn * 64 + j * 16 + lr) * 32 + half * 8;
            bf.h[0] = *(const bfx8*)(p);
            bf.h[1] = *(const bfx8*)(p + 16);
            #pragma unroll
            for (int mi = 0; mi < 2; ++mi)
                acc[mi][j] = __builtin_amdgcn_wmma_f32_16x16x32_bf16(
                    false, af[mi].v, false, bf.v, (short)0, acc[mi][j], false, false);
        }
        __syncthreads();
    }

    // ---- epilogue: 16x16 f32 C layout: lane<16 -> M=v, lane>=16 -> M=v+8, N=lr
    #pragma unroll
    for (int mi = 0; mi < 2; ++mi) {
        #pragma unroll
        for (int j = 0; j < 4; ++j) {
            const int c = nBase + wn * 64 + j * 16 + lr;
            if (c >= Nc) continue;
            #pragma unroll
            for (int v = 0; v < 8; ++v) {
                const int r = mBase + wm * 32 + mi * 16 + (half ? v + 8 : v);
                if (r >= M) continue;
                float x = acc[mi][j][v];
                if (mode == MODE_SIG) x = 1.0f / (1.0f + __expf(-x));
                if (mode == MODE_BF16) Cb[(long long)r * ldc + c] = (__bf16)x;
                else                   Cf[(long long)r * ldc + c] = x;
            }
        }
    }
}

// ---------------------------------------------------------------------------
// SpMM: O[row[e], :] += val[e] * S[col[e], :]   (S bf16 [N,512], O f32 [N,512])
// One wave per edge; lane covers 16 contiguous columns (2x b128 loads).
// ---------------------------------------------------------------------------
__global__ __launch_bounds__(256) void spmm_bf16(
    const int* __restrict__ row, const int* __restrict__ col,
    const float* __restrict__ val,
    const __bf16* __restrict__ S, float* __restrict__ O, long long E)
{
    const long long gid  = (long long)blockIdx.x * 256 + threadIdx.x;
    const long long e    = gid >> 5;
    const int       lane = (int)(gid & 31);
    if (e >= E) return;
    const int   r = row[e];
    const int   c = col[e];
    const float v = val[e];
    const __bf16* src = S + (long long)c * 512 + lane * 16;
    float*        dst = O + (long long)r * 512 + lane * 16;
    bfx8 x0 = *(const bfx8*)(src);
    bfx8 x1 = *(const bfx8*)(src + 8);
    #pragma unroll
    for (int i = 0; i < 8; ++i) atomicAdd(dst + i,     v * (float)x0[i]);
    #pragma unroll
    for (int i = 0; i < 8; ++i) atomicAdd(dst + 8 + i, v * (float)x1[i]);
}

// ---------------------------------------------------------------------------
// Elementwise / pack kernels
// ---------------------------------------------------------------------------
__global__ __launch_bounds__(256) void pack_x_kernel(
    const float* __restrict__ EH, const float* __restrict__ EI,
    const int* __restrict__ ind, __bf16* __restrict__ X, long long total)
{
    const long long t = (long long)blockIdx.x * 256 + threadIdx.x;
    if (t >= total) return;
    const long long n = t >> 9;
    const int       j = (int)(t & 511);
    const long long ii = ind[n];
    const float v = (j < 256) ? EH[ii * 256 + j] : EI[ii * 256 + (j - 256)];
    X[t] = (__bf16)v;
}

// Wt[j][k] for S = Xcat @ Wt^T, Xcat=[iH|iI], S=[sH|sI]
__global__ __launch_bounds__(256) void pack_w_kernel(
    const float* __restrict__ WH, const float* __restrict__ WI, __bf16* __restrict__ Wt)
{
    const int t = blockIdx.x * 256 + threadIdx.x;   // 512*512
    const int j = t >> 9, k = t & 511;
    float v;
    if (j < 256) v = (k < 256) ? WH[k * 256 + j] : -WI[(k - 256) * 256 + j];
    else {
        const int jp = j - 256;
        v = (k < 256) ? WI[k * 256 + jp] : WH[(k - 256) * 256 + jp];
    }
    Wt[t] = (__bf16)v;
}

__global__ __launch_bounds__(256) void bias_relu_pack(
    const float* __restrict__ O, const float* __restrict__ b1,
    const float* __restrict__ b2, __bf16* __restrict__ X, long long total)
{
    const long long t = (long long)blockIdx.x * 256 + threadIdx.x;
    if (t >= total) return;
    const int j = (int)(t & 511);
    const float bias = (j < 256) ? b1[j] : b2[j - 256];
    X[t] = (__bf16)fmaxf(O[t] + bias, 0.0f);
}

// fH = relu(oH + b1) + iH ; fI = relu(oI + b2) + iI  -> Fcat bf16 [N,512]
__global__ __launch_bounds__(256) void finalize_kernel(
    const float* __restrict__ O, const float* __restrict__ b1, const float* __restrict__ b2,
    const float* __restrict__ EH, const float* __restrict__ EI,
    const int* __restrict__ ind, __bf16* __restrict__ F, long long total)
{
    const long long t = (long long)blockIdx.x * 256 + threadIdx.x;
    if (t >= total) return;
    const long long n = t >> 9;
    const int       j = (int)(t & 511);
    const long long ii = ind[n];
    const float base = (j < 256) ? EH[ii * 256 + j] : EI[ii * 256 + j - 256];
    const float bias = (j < 256) ? b1[j] : b2[j - 256];
    F[t] = (__bf16)(fmaxf(O[t] + bias, 0.0f) + base);
}

__global__ __launch_bounds__(256) void gather_rel_kernel(
    const float* __restrict__ RH, const float* __restrict__ RI,
    const int* __restrict__ rel, __bf16* __restrict__ rh, __bf16* __restrict__ ri, int total)
{
    const int t = blockIdx.x * 256 + threadIdx.x;
    if (t >= total) return;
    const int b = t >> 8, j = t & 255;
    const long long rr = rel[b];
    rh[t] = (__bf16)RH[rr * 256 + j];
    ri[t] = (__bf16)RI[rr * 256 + j];
}

// WHJt[j][k] = j<256 ? WH[k][j] : WJ[k][j-256]; WIKt likewise with WI/WK.
__global__ __launch_bounds__(256) void pack_wquad_kernel(
    const float* __restrict__ WH, const float* __restrict__ WI,
    const float* __restrict__ WJ, const float* __restrict__ WK,
    __bf16* __restrict__ WHJt, __bf16* __restrict__ WIKt)
{
    const int t = blockIdx.x * 256 + threadIdx.x;   // 512*256
    const int j = t >> 8, k = t & 255;
    WHJt[t] = (__bf16)((j < 256) ? WH[k * 256 + j] : WJ[k * 256 + (j - 256)]);
    WIKt[t] = (__bf16)((j < 256) ? WI[k * 256 + j] : WK[k * 256 + (j - 256)]);
}

// ---------------------------------------------------------------------------
// BatchNorm helpers (training-mode, biased variance, over B=512 rows)
// ---------------------------------------------------------------------------
__device__ __forceinline__ float block_reduce_sum(float v, float* sd, int tid, int n)
{
    sd[tid] = v; __syncthreads();
    for (int s = n >> 1; s > 0; s >>= 1) {
        if (tid < s) sd[tid] += sd[tid + s];
        __syncthreads();
    }
    const float r = sd[0];
    __syncthreads();
    return r;
}

// grid = 512 blocks: (which<<8)|j ; block = 256 threads, each covers 2 batch rows
__global__ __launch_bounds__(256) void bn0_gather_kernel(
    const __bf16* __restrict__ Fcat, const int* __restrict__ bh,
    const float* __restrict__ g, const float* __restrict__ bta,
    float* __restrict__ xh, float* __restrict__ xi)
{
    __shared__ float sd[256];
    const int which = blockIdx.x >> 8;
    const int j     = blockIdx.x & 255;
    const int tid   = threadIdx.x;
    const int colo  = (which << 8) + j;
    const float v0 = (float)Fcat[(long long)bh[tid]       * 512 + colo];
    const float v1 = (float)Fcat[(long long)bh[tid + 256] * 512 + colo];
    const float s  = block_reduce_sum(v0 + v1, sd, tid, 256);
    const float s2 = block_reduce_sum(v0 * v0 + v1 * v1, sd, tid, 256);
    const float mu  = s * (1.0f / 512.0f);
    const float var = s2 * (1.0f / 512.0f) - mu * mu;
    const float rs  = rsqrtf(var + 1e-5f);
    const float gg = g[j], bb = bta[j];
    float* out = which ? xi : xh;
    out[tid * 256 + j]         = gg * (v0 - mu) * rs + bb;
    out[(tid + 256) * 256 + j] = gg * (v1 - mu) * rs + bb;
}

__device__ __forceinline__ float bn_col(float x, float* sd, int tid)
{
    const float s  = block_reduce_sum(x, sd, tid, 512);
    const float s2 = block_reduce_sum(x * x, sd, tid, 512);
    const float mu  = s * (1.0f / 512.0f);
    const float var = s2 * (1.0f / 512.0f) - mu * mu;
    return (x - mu) * rsqrtf(var + 1e-5f);
}

// grid = 256 (j), block = 512 (b). Builds Pcat = [a-d | b+c] in bf16.
__global__ __launch_bounds__(512) void mix_bn1_pack_kernel(
    const float* __restrict__ xh, const float* __restrict__ xi,
    const float* __restrict__ WhWj, const float* __restrict__ WiWk,
    const float* __restrict__ g, const float* __restrict__ bt,
    __bf16* __restrict__ Pcat)
{
    __shared__ float sd[512];
    const int j = blockIdx.x;
    const int b = threadIdx.x;
    const float h  = xh[b * 256 + j];
    const float im = xi[b * 256 + j];
    const float m1 = h  * WhWj[b * 512 + j];          // xh * Wh
    const float m2 = h  * WiWk[b * 512 + j];          // xh * Wi
    const float m3 = im * WhWj[b * 512 + 256 + j];    // xi * Wj
    const float m4 = im * WiWk[b * 512 + 256 + j];    // xi * Wk
    const float gg = g[j], bb = bt[j];
    const float a  = gg * bn_col(m1, sd, b) + bb;
    const float bo = gg * bn_col(m2, sd, b) + bb;
    const float c  = gg * bn_col(m3, sd, b) + bb;
    const float d  = gg * bn_col(m4, sd, b) + bb;
    Pcat[b * 512 + j]       = (__bf16)(a - d);
    Pcat[b * 512 + 256 + j] = (__bf16)(bo + c);
}

// ---------------------------------------------------------------------------
// Host launcher
// ---------------------------------------------------------------------------
extern "C" void kernel_launch(void* const* d_in, const int* in_sizes, int n_in,
                              void* d_out, int out_size, void* d_ws, size_t ws_size,
                              hipStream_t stream)
{
    const int D = 256;
    const int       N  = in_sizes[0] / D;
    const long long E  = in_sizes[20];
    const int       Bn = in_sizes[23];

    const float* EH    = (const float*)d_in[0];
    const float* EI    = (const float*)d_in[1];
    const float* RH    = (const float*)d_in[2];
    const float* RI    = (const float*)d_in[3];
    const float* WH    = (const float*)d_in[4];
    const float* WI    = (const float*)d_in[5];
    const float* WJ    = (const float*)d_in[6];
    const float* WK    = (const float*)d_in[7];
    const float* g1_WH = (const float*)d_in[8];
    const float* g1_WI = (const float*)d_in[9];
    const float* g1_b1 = (const float*)d_in[10];
    const float* g1_b2 = (const float*)d_in[11];
    const float* g2_WH = (const float*)d_in[12];
    const float* g2_WI = (const float*)d_in[13];
    const float* g2_b1 = (const float*)d_in[14];
    const float* g2_b2 = (const float*)d_in[15];
    const float* bn0_g = (const float*)d_in[16];
    const float* bn0_b = (const float*)d_in[17];
    const float* bn1_g = (const float*)d_in[18];
    const float* bn1_b = (const float*)d_in[19];
    const float* adj_v = (const float*)d_in[20];
    const int*   adj_r = (const int*)d_in[21];
    const int*   adj_c = (const int*)d_in[22];
    const int*   bhead = (const int*)d_in[23];
    const int*   brel  = (const int*)d_in[24];
    const int*   iind  = (const int*)d_in[25];

    // ---- workspace carve-out (256B aligned) ----
    char* wp = (char*)d_ws;
    auto take = [&](size_t bytes) {
        char* q = wp;
        wp += (bytes + 255) & ~(size_t)255;
        return q;
    };
    __bf16* Xcat = (__bf16*)take((size_t)N * 512 * sizeof(__bf16)); // later: Fcat
    __bf16* Sb   = (__bf16*)take((size_t)N * 512 * sizeof(__bf16));
    __bf16* Wt   = (__bf16*)take((size_t)512 * 512 * sizeof(__bf16));
    __bf16* WHJt = (__bf16*)take((size_t)512 * 256 * sizeof(__bf16));
    __bf16* WIKt = (__bf16*)take((size_t)512 * 256 * sizeof(__bf16));
    __bf16* rhb  = (__bf16*)take((size_t)Bn * 256 * sizeof(__bf16));
    __bf16* rib  = (__bf16*)take((size_t)Bn * 256 * sizeof(__bf16));
    float*  xh   = (float*)take((size_t)Bn * 256 * sizeof(float));
    float*  xi   = (float*)take((size_t)Bn * 256 * sizeof(float));
    float*  WhWj = (float*)take((size_t)Bn * 512 * sizeof(float));
    float*  WiWk = (float*)take((size_t)Bn * 512 * sizeof(float));
    __bf16* Pcat = (__bf16*)take((size_t)Bn * 512 * sizeof(__bf16));

    float* O = (float*)d_out;               // [N,512] spmm accumulator == out_size floats
    const long long totNX = (long long)N * 512;
    const int ewBlocks   = cdiv_i(totNX, 256);
    const int spmmBlocks = cdiv_i(E * 32, 256);

    // ======================= GCN layer 1 =======================
    pack_x_kernel<<<ewBlocks, 256, 0, stream>>>(EH, EI, iind, Xcat, totNX);
    pack_w_kernel<<<(512 * 512) / 256, 256, 0, stream>>>(g1_WH, g1_WI, Wt);
    wmma_gemm_nt<<<dim3(cdiv_i(512, 128), cdiv_i(N, 128)), 256, 0, stream>>>(
        Xcat, Wt, nullptr, Sb, N, 512, 512, 512, MODE_BF16);
    hipMemsetAsync(d_out, 0, (size_t)N * 512 * sizeof(float), stream);
    spmm_bf16<<<spmmBlocks, 256, 0, stream>>>(adj_r, adj_c, adj_v, Sb, O, E);
    bias_relu_pack<<<ewBlocks, 256, 0, stream>>>(O, g1_b1, g1_b2, Xcat, totNX);

    // ======================= GCN layer 2 =======================
    pack_w_kernel<<<(512 * 512) / 256, 256, 0, stream>>>(g2_WH, g2_WI, Wt);
    wmma_gemm_nt<<<dim3(cdiv_i(512, 128), cdiv_i(N, 128)), 256, 0, stream>>>(
        Xcat, Wt, nullptr, Sb, N, 512, 512, 512, MODE_BF16);
    hipMemsetAsync(d_out, 0, (size_t)N * 512 * sizeof(float), stream);
    spmm_bf16<<<spmmBlocks, 256, 0, stream>>>(adj_r, adj_c, adj_v, Sb, O, E);
    finalize_kernel<<<ewBlocks, 256, 0, stream>>>(O, g2_b1, g2_b2, EH, EI, iind, Xcat, totNX);
    // Xcat now holds Fcat = [fH | fI] in bf16

    // ======================= scoring head =======================
    gather_rel_kernel<<<cdiv_i((long long)Bn * 256, 256), 256, 0, stream>>>(
        RH, RI, brel, rhb, rib, Bn * 256);
    pack_wquad_kernel<<<(512 * 256) / 256, 256, 0, stream>>>(WH, WI, WJ, WK, WHJt, WIKt);
    wmma_gemm_nt<<<dim3(cdiv_i(512, 128), cdiv_i(Bn, 128)), 256, 0, stream>>>(
        rhb, WHJt, WhWj, nullptr, Bn, 512, 256, 512, MODE_F32);   // [Wh | Wj]
    wmma_gemm_nt<<<dim3(cdiv_i(512, 128), cdiv_i(Bn, 128)), 256, 0, stream>>>(
        rib, WIKt, WiWk, nullptr, Bn, 512, 256, 512, MODE_F32);   // [Wi | Wk]
    bn0_gather_kernel<<<512, 256, 0, stream>>>(Xcat, bhead, bn0_g, bn0_b, xh, xi);
    mix_bn1_pack_kernel<<<256, 512, 0, stream>>>(xh, xi, WhWj, WiWk, bn1_g, bn1_b, Pcat);

    // score = sigmoid(Pcat @ Fcat^T) -> d_out [Bn, N]
    wmma_gemm_nt<<<dim3(cdiv_i(N, 128), cdiv_i(Bn, 128)), 256, 0, stream>>>(
        Pcat, Xcat, (float*)d_out, nullptr, Bn, N, 512, N, MODE_SIG);
}
